// MoEMLP_5308579578134
// MI455X (gfx1250) — matrix operand order repaired and checked by reading, not compiled, not run on hardware
//
#include <hip/hip_runtime.h>
#include <stdint.h>

// Problem constants (match reference)
#define Bc 4
#define Sc 2048
#define Dc 1024
#define Ec 8
#define Hc 1024
#define Kc 2
#define Tc (Bc * Sc)      // 8192 tokens
#define EH (Ec * Hc)      // 8192

// ---- CDNA5 WMMA types ----
typedef __attribute__((ext_vector_type(16))) __bf16 v16bf;
typedef __attribute__((ext_vector_type(8)))  float  v8f;

union BFrag { uint32_t u[8]; v16bf v; };
union FAcc  { v8f v; float f[8]; };

__device__ __forceinline__ uint32_t bfbits(float f) {
  union { float f; uint32_t u; } a; a.f = f;
  uint32_t u = a.u;
  uint32_t r = u + 0x7FFFu + ((u >> 16) & 1u);   // round-to-nearest-even
  return r >> 16;
}
__device__ __forceinline__ uint32_t packbf(float lo, float hi) {
  return bfbits(lo) | (bfbits(hi) << 16);
}

// ---- CDNA5 async global->LDS copy (ASYNCcnt-tracked), with fallback ----
#define AS1 __attribute__((address_space(1)))
#define AS3 __attribute__((address_space(3)))
typedef int v4i_ls __attribute__((vector_size(16)));   // int4, per builtin signature

__device__ __forceinline__ void async_copy16(const void* gptr, void* lptr) {
#if __has_builtin(__builtin_amdgcn_global_load_async_to_lds_b128)
  __builtin_amdgcn_global_load_async_to_lds_b128(
      (AS1 v4i_ls*)gptr, (AS3 v4i_ls*)lptr, 0, 0);
#else
  const uint32_t* g = (const uint32_t*)gptr;
  uint32_t* l = (uint32_t*)lptr;
  l[0] = g[0]; l[1] = g[1]; l[2] = g[2]; l[3] = g[3];
#endif
}
__device__ __forceinline__ void async_wait0() {
#if __has_builtin(__builtin_amdgcn_s_wait_asynccnt)
  __builtin_amdgcn_s_wait_asynccnt(0);
#endif
}

// ---- Workspace layout (bytes) ----
// [0,1024): counters/stats (zeroed each call)
//   +0 cnt[E] (int), +64 offs[E+1] (int), +128 stats[2] (float),
//   +256 pseq[B*E] (float), +512 fseq[B*E] (float)
#define WS_PERMC 1024u                          // E*T ints      (256 KB)
#define WS_PWTC  (WS_PERMC + Ec * Tc * 4u)      // E*T floats    (256 KB)
#define WS_PERM  (WS_PWTC + Ec * Tc * 4u)       // T*K ints      (64 KB)
#define WS_PWT   (WS_PERM + Tc * Kc * 4u)       // T*K floats    (64 KB)
#define WS_XB    (WS_PWT + Tc * Kc * 4u)        // T*D bf16      (16 MB)
#define WS_W1T   (WS_XB + (size_t)Tc * Dc * 2u) // EH*D bf16 (transposed w1)
#define WS_W2T   (WS_W1T + (size_t)EH * Dc * 2u)// D*EH bf16 (transposed w2)
#define WS_HBUF  (WS_W2T + (size_t)Dc * EH * 2u)// T*K*H bf16    (32 MB)

// =====================================================================
// One-time (per call) bf16 conversion / transposition passes
// =====================================================================
__global__ __launch_bounds__(256) void conv_x(const float* __restrict__ x,
                                              uint32_t* __restrict__ xb32) {
  size_t i = ((size_t)blockIdx.x * 256 + threadIdx.x) * 4;  // dword pairs
  #pragma unroll
  for (int j = 0; j < 4; j++)
    xb32[i + j] = packbf(x[2 * (i + j)], x[2 * (i + j) + 1]);
}

// w1t[col][d] = bf16(w1[d][col]) ; col in [0,EH), d in [0,D)
__global__ __launch_bounds__(256) void conv_w1t(const float* __restrict__ w1,
                                                uint16_t* __restrict__ w1t) {
  int col = blockIdx.x * 256 + threadIdx.x;
  int d0  = blockIdx.y * 128;
  for (int d = d0; d < d0 + 128; d++)
    w1t[(size_t)col * Dc + d] = (uint16_t)bfbits(w1[(size_t)d * EH + col]);
}

// w2t[col][h] = bf16(w2[h][col]) ; col in [0,D), h in [0,EH)
__global__ __launch_bounds__(256) void conv_w2t(const float* __restrict__ w2,
                                                uint16_t* __restrict__ w2t) {
  int col = blockIdx.x * 256 + threadIdx.x;
  int h0  = blockIdx.y * 128;
  for (int h = h0; h < h0 + 128; h++)
    w2t[(size_t)col * EH + h] = (uint16_t)bfbits(w2[(size_t)h * Dc + col]);
}

// =====================================================================
// Router: one wave (32 lanes) per token.
// =====================================================================
__global__ __launch_bounds__(256) void moe_router(
    const float* __restrict__ x, const float* __restrict__ rw,
    const float* __restrict__ ebias,
    int* __restrict__ cnt, int* __restrict__ permc, float* __restrict__ pwtc,
    float* __restrict__ stats, float* __restrict__ pseq, float* __restrict__ fseq)
{
  int wid  = threadIdx.x >> 5;
  int lane = threadIdx.x & 31;
  int t    = blockIdx.x * 8 + wid;
  if (t >= Tc) return;

  const float* xr = x + (size_t)t * Dc;
  float acc[Ec];
  #pragma unroll
  for (int e = 0; e < Ec; e++) acc[e] = 0.f;

  for (int i = lane; i < Dc; i += 32) {
    float xv = xr[i];
    #pragma unroll
    for (int e = 0; e < Ec; e++) acc[e] += xv * rw[e * Dc + i];
  }
  #pragma unroll
  for (int e = 0; e < Ec; e++) {
    #pragma unroll
    for (int o = 16; o > 0; o >>= 1) acc[e] += __shfl_xor(acc[e], o, 32);
  }

  if (lane == 0) {
    float mx = acc[0];
    #pragma unroll
    for (int e = 1; e < Ec; e++) mx = acc[e] > mx ? acc[e] : mx;
    float se = 0.f;
    #pragma unroll
    for (int e = 0; e < Ec; e++) se += __expf(acc[e] - mx);
    float lse = mx + __logf(se);
    atomicAdd(&stats[0], lse * lse);

    int b = t / Sc;
    float p[Ec], s[Ec], sp = 0.f;
    #pragma unroll
    for (int e = 0; e < Ec; e++) {
      p[e] = 1.f / (1.f + __expf(-acc[e]));
      sp += p[e];
      s[e] = p[e] + ebias[e];
      atomicAdd(&pseq[b * Ec + e], p[e]);
    }
    atomicAdd(&stats[1], sp);

    int i1 = 0;
    #pragma unroll
    for (int e = 1; e < Ec; e++) if (s[e] > s[i1]) i1 = e;
    int i2 = (i1 == 0) ? 1 : 0;
    #pragma unroll
    for (int e = 0; e < Ec; e++) if (e != i1 && s[e] > s[i2]) i2 = e;

    float wsum = p[i1] + p[i2] + 1e-20f;
    float wa = p[i1] / wsum, wb = p[i2] / wsum;

    int pos1 = atomicAdd(&cnt[i1], 1);
    permc[i1 * Tc + pos1] = t;  pwtc[i1 * Tc + pos1] = wa;
    int pos2 = atomicAdd(&cnt[i2], 1);
    permc[i2 * Tc + pos2] = t;  pwtc[i2 * Tc + pos2] = wb;

    atomicAdd(&fseq[b * Ec + i1], 1.f);
    atomicAdd(&fseq[b * Ec + i2], 1.f);
  }
}

__global__ void moe_scan(const int* __restrict__ cnt, int* __restrict__ offs) {
  if (threadIdx.x == 0 && blockIdx.x == 0) {
    int a = 0;
    for (int e = 0; e < Ec; e++) { offs[e] = a; a += cnt[e]; }
    offs[Ec] = a;
  }
}

__global__ __launch_bounds__(256) void moe_compact(
    const int* __restrict__ cnt, const int* __restrict__ offs,
    const int* __restrict__ permc, const float* __restrict__ pwtc,
    int* __restrict__ perm, float* __restrict__ pwt)
{
  int e = blockIdx.x;
  int c = cnt[e], o = offs[e];
  for (int i = threadIdx.x; i < c; i += blockDim.x) {
    perm[o + i] = permc[e * Tc + i];
    pwt[o + i]  = pwtc[e * Tc + i];
  }
}

// =====================================================================
// Grouped GEMMs. Block tile 128x128, K-step 32, 8 waves of 64x32.
// All operands bf16, staged into LDS with async b128 copies.
// LDS pitches are 20 dwords so every b128 target is 16B-aligned.
// =====================================================================
#define LDA 20
#define LDB 20

__global__ __launch_bounds__(256) void moe_gemm1(
    const uint16_t* __restrict__ xb, const uint16_t* __restrict__ w1t,
    const int* __restrict__ perm, const float* __restrict__ pwt,
    const int* __restrict__ cnt, const int* __restrict__ offs,
    uint16_t* __restrict__ hbuf)
{
  __shared__ __align__(16) uint32_t Alds[128 * LDA];
  __shared__ __align__(16) uint32_t Blds[128 * LDB];
  __shared__ int   tokLds[128];
  __shared__ float wLds[128];

  int e = blockIdx.z;
  int cn = cnt[e];
  int mbase = blockIdx.y * 128;
  if (mbase >= cn) return;
  int off = offs[e];
  int rem = cn - mbase; if (rem > 128) rem = 128;

  int tid = threadIdx.x;
  if (tid < 128) {
    int rr = (tid < rem) ? tid : 0;
    tokLds[tid] = perm[off + mbase + rr];
    wLds[tid]   = pwt[off + mbase + rr];
  }
  __syncthreads();

  // A-loader: row = tid/2, 16 consecutive K (two b128s)
  int ar = tid >> 1;
  int kl = (tid & 1) << 4;                      // 0 or 16
  const uint16_t* xrow = xb + (size_t)tokLds[ar] * Dc;
  uint32_t* aDst = &Alds[ar * LDA + (kl >> 1)];

  // B-loader: one column of w1 (row of the transposed matrix)
  int bn  = tid & 127;
  int bkb = (tid >> 7) << 4;                    // 0 or 16
  const uint16_t* brow = w1t + (size_t)(e * Hc + blockIdx.x * 128 + bn) * Dc;
  uint32_t* bDst = &Blds[bn * LDB + (bkb >> 1)];

  // wave tiling
  int lane = tid & 31, wid = tid >> 5;
  int wm = (wid >> 2) * 64;
  int wn = (wid & 3) * 32;
  int fm  = lane & 15;
  int akb = (lane < 16) ? 0 : 4;
  int bkd = (lane < 16) ? 0 : 4;

  FAcc acc[4][2];
  #pragma unroll
  for (int i = 0; i < 4; i++)
    #pragma unroll
    for (int j = 0; j < 2; j++)
      #pragma unroll
      for (int v = 0; v < 8; v++) acc[i][j].f[v] = 0.f;

  for (int kg = 0; kg < Dc; kg += 32) {
    __syncthreads();
    async_copy16(xrow + kg + kl,      aDst);
    async_copy16(xrow + kg + kl + 8,  aDst + 4);
    async_copy16(brow + kg + bkb,     bDst);
    async_copy16(brow + kg + bkb + 8, bDst + 4);
    async_wait0();
    __syncthreads();

    BFrag a[4], bf[2];
    #pragma unroll
    for (int i = 0; i < 4; i++) {
      int base = (wm + i * 16 + fm) * LDA + akb;
      #pragma unroll
      for (int j = 0; j < 4; j++) { a[i].u[j] = Alds[base + j]; a[i].u[4 + j] = Alds[base + 8 + j]; }
    }
    #pragma unroll
    for (int j = 0; j < 2; j++) {
      int base = (wn + j * 16 + fm) * LDB + bkd;
      #pragma unroll
      for (int q = 0; q < 4; q++) { bf[j].u[q] = Blds[base + q]; bf[j].u[4 + q] = Blds[base + 8 + q]; }
    }
    #pragma unroll
    for (int i = 0; i < 4; i++)
      #pragma unroll
      for (int j = 0; j < 2; j++)
        acc[i][j].v = __builtin_amdgcn_wmma_f32_16x16x32_bf16(
            false, a[i].v, false, bf[j].v, (short)0, acc[i][j].v, false, false);
  }

  // epilogue: relu^2 * route weight -> bf16 hbuf
  int nsel = lane & 15;
  int mhi  = (lane < 16) ? 0 : 8;
  #pragma unroll
  for (int i = 0; i < 4; i++)
    #pragma unroll
    for (int j = 0; j < 2; j++) {
      int nloc = wn + j * 16 + nsel;
      size_t hcol = (size_t)blockIdx.x * 128 + nloc;
      #pragma unroll
      for (int v = 0; v < 8; v++) {
        int mloc = wm + i * 16 + v + mhi;
        if (mloc < rem) {
          float sv = acc[i][j].f[v];
          sv = sv > 0.f ? sv * sv : 0.f;
          sv *= wLds[mloc];
          hbuf[(size_t)(off + mbase + mloc) * Hc + hcol] = (uint16_t)bfbits(sv);
        }
      }
    }
}

__global__ __launch_bounds__(256) void moe_gemm2(
    const uint16_t* __restrict__ hbuf, const uint16_t* __restrict__ w2t,
    const int* __restrict__ perm, const int* __restrict__ cnt,
    const int* __restrict__ offs, float* __restrict__ out)
{
  __shared__ __align__(16) uint32_t Alds[128 * LDA];
  __shared__ __align__(16) uint32_t Blds[128 * LDB];
  __shared__ int tokLds[128];

  int e = blockIdx.z;
  int cn = cnt[e];
  int mbase = blockIdx.y * 128;
  if (mbase >= cn) return;
  int off = offs[e];
  int rem = cn - mbase; if (rem > 128) rem = 128;

  int tid = threadIdx.x;
  if (tid < 128) {
    int rr = (tid < rem) ? tid : 0;
    tokLds[tid] = perm[off + mbase + rr];
  }
  __syncthreads();

  int ar = tid >> 1;
  int kl = (tid & 1) << 4;
  int arr = (ar < rem) ? ar : 0;
  const uint16_t* hrow = hbuf + (size_t)(off + mbase + arr) * Hc;
  uint32_t* aDst = &Alds[ar * LDA + (kl >> 1)];

  int bn  = tid & 127;
  int bkb = (tid >> 7) << 4;
  const uint16_t* brow = w2t + (size_t)(blockIdx.x * 128 + bn) * EH + (size_t)e * Hc;
  uint32_t* bDst = &Blds[bn * LDB + (bkb >> 1)];

  int lane = tid & 31, wid = tid >> 5;
  int wm = (wid >> 2) * 64;
  int wn = (wid & 3) * 32;
  int fm  = lane & 15;
  int akb = (lane < 16) ? 0 : 4;
  int bkd = (lane < 16) ? 0 : 4;

  FAcc acc[4][2];
  #pragma unroll
  for (int i = 0; i < 4; i++)
    #pragma unroll
    for (int j = 0; j < 2; j++)
      #pragma unroll
      for (int v = 0; v < 8; v++) acc[i][j].f[v] = 0.f;

  for (int kg = 0; kg < Hc; kg += 32) {
    __syncthreads();
    async_copy16(hrow + kg + kl,      aDst);
    async_copy16(hrow + kg + kl + 8,  aDst + 4);
    async_copy16(brow + kg + bkb,     bDst);
    async_copy16(brow + kg + bkb + 8, bDst + 4);
    async_wait0();
    __syncthreads();

    BFrag a[4], bf[2];
    #pragma unroll
    for (int i = 0; i < 4; i++) {
      int base = (wm + i * 16 + fm) * LDA + akb;
      #pragma unroll
      for (int j = 0; j < 4; j++) { a[i].u[j] = Alds[base + j]; a[i].u[4 + j] = Alds[base + 8 + j]; }
    }
    #pragma unroll
    for (int j = 0; j < 2; j++) {
      int base = (wn + j * 16 + fm) * LDB + bkd;
      #pragma unroll
      for (int q = 0; q < 4; q++) { bf[j].u[q] = Blds[base + q]; bf[j].u[4 + q] = Blds[base + 8 + q]; }
    }
    #pragma unroll
    for (int i = 0; i < 4; i++)
      #pragma unroll
      for (int j = 0; j < 2; j++)
        acc[i][j].v = __builtin_amdgcn_wmma_f32_16x16x32_bf16(
            false, a[i].v, false, bf[j].v, (short)0, acc[i][j].v, false, false);
  }

  int nsel = lane & 15;
  int mhi  = (lane < 16) ? 0 : 8;
  #pragma unroll
  for (int i = 0; i < 4; i++)
    #pragma unroll
    for (int j = 0; j < 2; j++) {
      int nloc = wn + j * 16 + nsel;
      size_t col = (size_t)blockIdx.x * 128 + nloc;
      #pragma unroll
      for (int v = 0; v < 8; v++) {
        int mloc = wm + i * 16 + v + mhi;
        if (mloc < rem) {
          int tok = tokLds[mloc];
          atomicAdd(&out[(size_t)tok * Dc + col], acc[i][j].f[v]);
        }
      }
    }
}

// =====================================================================
__global__ void moe_finalize(const int* __restrict__ cnt,
                             const float* __restrict__ stats,
                             const float* __restrict__ pseq,
                             const float* __restrict__ fseq,
                             float* __restrict__ outTail)
{
  if (threadIdx.x == 0 && blockIdx.x == 0) {
    float z  = stats[0] / (float)Tc;
    float cl = stats[1] / (float)Tc;
    float lb = 0.f;
    for (int b = 0; b < Bc; b++) {
      float s = 0.f;
      for (int e = 0; e < Ec; e++)
        s += (fseq[b * Ec + e] / (float)(Sc * Kc)) * (pseq[b * Ec + e] / (float)Sc);
      lb += s;
    }
    lb = (float)Ec * lb / (float)Bc;
    outTail[0] = z;
    outTail[1] = lb;
    outTail[2] = cl;
    for (int e = 0; e < Ec; e++)
      outTail[3 + e] = (float)cnt[e] / (float)(Tc * Kc);
  }
}

// =====================================================================
extern "C" void kernel_launch(void* const* d_in, const int* in_sizes, int n_in,
                              void* d_out, int out_size, void* d_ws, size_t ws_size,
                              hipStream_t stream) {
  const float* x   = (const float*)d_in[0];
  const float* rw  = (const float*)d_in[1];
  const float* w1  = (const float*)d_in[2];
  const float* w2  = (const float*)d_in[3];
  const float* eb  = (const float*)d_in[4];
  float* out = (float*)d_out;

  char* ws = (char*)d_ws;
  int*   cnt   = (int*)(ws + 0);
  int*   offs  = (int*)(ws + 64);
  float* stats = (float*)(ws + 128);
  float* pseq  = (float*)(ws + 256);
  float* fseq  = (float*)(ws + 512);
  int*   permc = (int*)(ws + WS_PERMC);
  float* pwtc  = (float*)(ws + WS_PWTC);
  int*   perm  = (int*)(ws + WS_PERM);
  float* pwt   = (float*)(ws + WS_PWT);
  uint16_t* xb   = (uint16_t*)(ws + WS_XB);
  uint16_t* w1t  = (uint16_t*)(ws + WS_W1T);
  uint16_t* w2t  = (uint16_t*)(ws + WS_W2T);
  uint16_t* hbuf = (uint16_t*)(ws + WS_HBUF);

  (void)hipMemsetAsync(ws, 0, 1024, stream);
  (void)hipMemsetAsync(out, 0, (size_t)Tc * Dc * sizeof(float), stream);

  // one-time bf16 conversion / transposition
  conv_x<<<(Tc * Dc) / 2048, 256, 0, stream>>>(x, (uint32_t*)xb);
  conv_w1t<<<dim3(EH / 256, Dc / 128), 256, 0, stream>>>(w1, w1t);
  conv_w2t<<<dim3(Dc / 256, EH / 128), 256, 0, stream>>>(w2, w2t);

  moe_router<<<Tc / 8, 256, 0, stream>>>(x, rw, eb, cnt, permc, pwtc, stats, pseq, fseq);
  moe_scan<<<1, 32, 0, stream>>>(cnt, offs);
  moe_compact<<<Ec, 256, 0, stream>>>(cnt, offs, permc, pwtc, perm, pwt);

  moe_gemm1<<<dim3(Hc / 128, Tc / 128, Ec), 256, 0, stream>>>(
      xb, w1t, perm, pwt, cnt, offs, hbuf);
  moe_gemm2<<<dim3(Dc / 128, Tc / 128, Ec), 256, 0, stream>>>(
      hbuf, w2t, perm, cnt, offs, out);

  moe_finalize<<<1, 32, 0, stream>>>(cnt, stats, pseq, fseq, out + (size_t)Tc * Dc);
}
